// TopkMoE_60997125538169
// MI455X (gfx1250) — matrix operand order repaired and checked by reading, not compiled, not run on hardware
//
#include <hip/hip_runtime.h>

// ---------------------------------------------------------------------------
// Top-2 MoE forward for MI455X (gfx1250, wave32, WMMA).
//
// Routed: per token only its 2 selected experts run. ~137 GFLOP vs ~285 MB of
// mandatory traffic -> compute bound -> feed the matrix pipe.
// fp32 math is done as bf16 hi/lo split (3x v_wmma_f32_16x16x32_bf16, f32 acc).
//
// FAST path (ws >= ~420 MB): one-time conversion kernels emit
//   - x / h as natural-layout bf16 hi/lo planes (A staging = pure b128 copies)
//   - w1 / w2 as B-FRAGMENT-TILED bf16 hi/lo planes: per (n-tile, k-chunk) a
//     1 KB block of [lane][16 elems] -> B fragments load straight from global
//     as one aligned 32B vector per plane (no LDS B tile, no shuffles).
//   Inner loop: ds_load_b128 A-fragments + global b128 B-fragments + 24 WMMA.
// FALLBACK path (ws >= ~135 MB): conversion inside the GEMM staging (round-2
//   version), B via strided u16 LDS reads.
// ---------------------------------------------------------------------------

#define T_TOK 4096
#define DIMX  1024
#define HIDX  4096
#define NEXP  8

#define BLDS  136   // fallback B row stride (bf16): 272B rows, aligned b128

typedef __attribute__((ext_vector_type(16))) __bf16 v16bf;
typedef __attribute__((ext_vector_type(8)))  __bf16 v8bf;
typedef __attribute__((ext_vector_type(8)))  float  v8f;

// ---- fragment helpers ------------------------------------------------------

// A fragment (16x32 bf16, ISA 7.12.2): LDS already in fragment order,
// lane L's 16 elements contiguous at (mt*32 + L)*16.
__device__ __forceinline__ void load_a_frags(const __bf16* __restrict__ sAhi,
                                             const __bf16* __restrict__ sAlo,
                                             int mt, int lane,
                                             v16bf& ahi, v16bf& alo) {
  const int off = (mt * 32 + lane) * 16;     // 32B-aligned
  ahi = *(const v16bf*)(sAhi + off);
  alo = *(const v16bf*)(sAlo + off);
}

// Fallback B fragment (32x16 bf16) from [k][n] LDS: lane -> col n0+(L&15),
// K = 16*(L>>4)+e (strided u16 reads).
__device__ __forceinline__ void load_b_frags_lds(const __bf16* __restrict__ sBhi,
                                                 const __bf16* __restrict__ sBlo,
                                                 int n0, int lane,
                                                 v16bf& bhi, v16bf& blo) {
  const int n = n0 + (lane & 15);
  const int kb = (lane >> 4) * 16;
  const __bf16* ph = sBhi + kb * BLDS + n;
  const __bf16* pl = sBlo + kb * BLDS + n;
#pragma unroll
  for (int e = 0; e < 16; ++e) {
    bhi[e] = ph[e * BLDS];
    blo[e] = pl[e * BLDS];
  }
}

__device__ __forceinline__ v8f wmma3(v16bf ahi, v16bf alo, v16bf bhi, v16bf blo, v8f c) {
  c = __builtin_amdgcn_wmma_f32_16x16x32_bf16(false, ahi, false, bhi, (short)0, c, false, false);
  c = __builtin_amdgcn_wmma_f32_16x16x32_bf16(false, ahi, false, blo, (short)0, c, false, false);
  c = __builtin_amdgcn_wmma_f32_16x16x32_bf16(false, alo, false, bhi, (short)0, c, false, false);
  return c;
}

// Convert 8 consecutive fp32 to bf16 hi/lo vectors.
__device__ __forceinline__ void cvt8(const float* __restrict__ src, v8bf& hv, v8bf& lv) {
  const float4 f0 = *(const float4*)(src);
  const float4 f1 = *(const float4*)(src + 4);
  float f[8] = {f0.x, f0.y, f0.z, f0.w, f1.x, f1.y, f1.z, f1.w};
#pragma unroll
  for (int j = 0; j < 8; ++j) {
    const __bf16 h = (__bf16)f[j];
    hv[j] = h;
    lv[j] = (__bf16)(f[j] - (float)h);
  }
}

// A staging swizzle: 8 consecutive K of row m -> 8 consecutive fragment elems
// of lane (m&15)+16*(k8>>1), element base (k8&1)*8.
#define A_DST(m, k8) ((((m) >> 4) * 32 + ((m) & 15) + 16 * ((k8) >> 1)) * 16 + ((k8) & 1) * 8)

// ---------------------------------------------------------------------------
// 0) zero per-expert counts
// ---------------------------------------------------------------------------
__global__ void init_cnt_kernel(int* __restrict__ cnt) {
  if (threadIdx.x < NEXP) cnt[threadIdx.x] = 0;
}

// ---------------------------------------------------------------------------
// 1) gating
// ---------------------------------------------------------------------------
__global__ void __launch_bounds__(256)
gate_kernel(const float* __restrict__ x, const float* __restrict__ gw,
            const float* __restrict__ gb, int* __restrict__ cnt,
            int* __restrict__ pairlist, int* __restrict__ topk_e,
            float* __restrict__ topk_p) {
  __shared__ float sgw[DIMX * NEXP];
  const int tid = threadIdx.x;
  for (int i = tid; i < DIMX * NEXP; i += 256) sgw[i] = gw[i];
  __syncthreads();

  const int t = blockIdx.x * 256 + tid;
  const float* xr = x + (size_t)t * DIMX;
  float acc[NEXP];
#pragma unroll
  for (int e = 0; e < NEXP; ++e) acc[e] = gb[e];
  for (int k = 0; k < DIMX; k += 4) {
    const float4 xv = *(const float4*)(xr + k);
#pragma unroll
    for (int e = 0; e < NEXP; ++e) {
      acc[e] += xv.x * sgw[(k + 0) * NEXP + e] + xv.y * sgw[(k + 1) * NEXP + e]
              + xv.z * sgw[(k + 2) * NEXP + e] + xv.w * sgw[(k + 3) * NEXP + e];
    }
  }
  float m = acc[0];
#pragma unroll
  for (int e = 1; e < NEXP; ++e) m = fmaxf(m, acc[e]);
  float p[NEXP], s = 0.f;
#pragma unroll
  for (int e = 0; e < NEXP; ++e) { p[e] = __expf(acc[e] - m); s += p[e]; }
  const float inv = 1.f / s;
#pragma unroll
  for (int e = 0; e < NEXP; ++e) p[e] *= inv;

  int e0 = 0;
#pragma unroll
  for (int e = 1; e < NEXP; ++e) if (p[e] > p[e0]) e0 = e;
  int e1 = (e0 == 0) ? 1 : 0;
#pragma unroll
  for (int e = 0; e < NEXP; ++e) if (e != e0 && p[e] > p[e1]) e1 = e;

  topk_e[t * 2 + 0] = e0;  topk_p[t * 2 + 0] = p[e0];
  topk_e[t * 2 + 1] = e1;  topk_p[t * 2 + 1] = p[e1];
  const int pos0 = atomicAdd(&cnt[e0], 1);
  pairlist[e0 * T_TOK + pos0] = t * 2 + 0;
  const int pos1 = atomicAdd(&cnt[e1], 1);
  pairlist[e1 * T_TOK + pos1] = t * 2 + 1;
}

// ---------------------------------------------------------------------------
// 2) out = comb @ b2
// ---------------------------------------------------------------------------
__global__ void __launch_bounds__(256)
out_init_kernel(const float* __restrict__ b2, const int* __restrict__ topk_e,
                const float* __restrict__ topk_p, float* __restrict__ out) {
  const int t  = blockIdx.x;
  const int e0 = topk_e[t * 2 + 0], e1 = topk_e[t * 2 + 1];
  const float p0 = topk_p[t * 2 + 0], p1 = topk_p[t * 2 + 1];
  const int d = threadIdx.x * 4;
  const float4 a = *(const float4*)(b2 + (size_t)e0 * DIMX + d);
  const float4 b = *(const float4*)(b2 + (size_t)e1 * DIMX + d);
  float4 o;
  o.x = p0 * a.x + p1 * b.x;
  o.y = p0 * a.y + p1 * b.y;
  o.z = p0 * a.z + p1 * b.z;
  o.w = p0 * a.w + p1 * b.w;
  *(float4*)(out + (size_t)t * DIMX + d) = o;
}

// ===========================================================================
// FAST PATH: one-time conversions
// ===========================================================================

// x (or any row-major fp32 matrix) -> natural-layout bf16 hi/lo planes.
__global__ void __launch_bounds__(256)
cvt_x_kernel(const float* __restrict__ x, __bf16* __restrict__ xh,
             __bf16* __restrict__ xl) {
  const size_t base = ((size_t)blockIdx.x * 256 + threadIdx.x) * 8;
  v8bf hv, lv;
  cvt8(x + base, hv, lv);
  *(v8bf*)(xh + base) = hv;
  *(v8bf*)(xl + base) = lv;
}

// Weights [E][K][N] fp32 -> B-fragment-tiled bf16 planes:
//   dst[(((e*NT + nt)*KC + kc)*32 + lane)*16 + elem],
//   lane = (n&15)+16*((k>>4)&1), elem = k&15, nt=n>>4, kc=k>>5.
// One thread: 16 consecutive k at fixed n -> one v16bf (32B) store per plane.
// Reads are wave-coalesced (consecutive n per row), stores are dense 32B.
__global__ void __launch_bounds__(256)
cvt_w_kernel(const float* __restrict__ w, __bf16* __restrict__ dhi,
             __bf16* __restrict__ dlo, int K, int N) {
  const int g   = blockIdx.x * 256 + threadIdx.x;  // (e, k16, n)
  const int n   = g % N;
  const int r   = g / N;
  const int k16 = r % (K / 16);
  const int e   = r / (K / 16);
  const int k0  = k16 * 16;
  const int kc  = k0 >> 5;
  const int h   = (k0 >> 4) & 1;
  const int nt  = n >> 4;
  const int lane = (n & 15) + 16 * h;
  const int NT = N / 16, KC = K / 32;
  const size_t off = ((((size_t)(e * NT + nt)) * KC + kc) * 32 + lane) * 16;
  const float* src = w + ((size_t)e * K + k0) * N + n;
  v16bf hv, lv;
#pragma unroll
  for (int i = 0; i < 16; ++i) {
    const float v = src[(size_t)i * N];
    const __bf16 hb = (__bf16)v;
    hv[i] = hb;
    lv[i] = (__bf16)(v - (float)hb);
  }
  *(v16bf*)(dhi + off) = hv;
  *(v16bf*)(dlo + off) = lv;
}

// ---------------------------------------------------------------------------
// FAST FFN1: h(hi/lo)[pid, n] = p * relu( x[tok] @ w1[e][:,n] + b1[e][n] )
// ---------------------------------------------------------------------------
__global__ void __launch_bounds__(256)
ffn1_fast_kernel(const __bf16* __restrict__ xh, const __bf16* __restrict__ xl,
                 const __bf16* __restrict__ w1fh, const __bf16* __restrict__ w1fl,
                 const float* __restrict__ b1, const int* __restrict__ cnt,
                 const int* __restrict__ pairlist, const float* __restrict__ topk_p,
                 __bf16* __restrict__ hh, __bf16* __restrict__ hl) {
  const int e  = blockIdx.z;
  const int ne = cnt[e];
  const int m0 = blockIdx.x * 128;
  if (m0 >= ne) return;                     // block-uniform: EXEC stays full
  const int n0 = blockIdx.y * 128;

  __shared__ __attribute__((aligned(32))) __bf16 sAhi[128 * 32];
  __shared__ __attribute__((aligned(32))) __bf16 sAlo[128 * 32];
  __shared__ int srow[128];

  const int tid = threadIdx.x, lane = tid & 31, wave = tid >> 5;
  for (int i = tid; i < 128; i += 256)
    srow[i] = pairlist[e * T_TOK + ((m0 + i) < ne ? (m0 + i) : 0)];
  __syncthreads();

  const v8f zero = {0.f, 0.f, 0.f, 0.f, 0.f, 0.f, 0.f, 0.f};
  v8f c[8];
#pragma unroll
  for (int i = 0; i < 8; ++i) c[i] = zero;

  const int NT = HIDX / 16, KC = DIMX / 32;
  const size_t bb = ((((size_t)(e * NT + (n0 >> 4) + wave)) * KC) * 32 + lane) * 16;

  for (int kc = 0; kc < KC; ++kc) {
    const int k0 = kc * 32;
    // stage A: pure bf16 copies into fragment-ordered LDS
    for (int i = tid; i < 512; i += 256) {
      const int m = i >> 2, k8 = i & 3;
      const size_t rb = (size_t)(srow[m] >> 1) * DIMX + k0 + k8 * 8;
      const int dst = A_DST(m, k8);
      *(v8bf*)(sAhi + dst) = *(const v8bf*)(xh + rb);
      *(v8bf*)(sAlo + dst) = *(const v8bf*)(xl + rb);
    }
    __syncthreads();
    // B fragments straight from fragment-tiled global (32B/lane, coalesced)
    const v16bf bhi = *(const v16bf*)(w1fh + bb + (size_t)kc * 512);
    const v16bf blo = *(const v16bf*)(w1fl + bb + (size_t)kc * 512);
#pragma unroll
    for (int mt = 0; mt < 8; ++mt) {
      v16bf ahi, alo;
      load_a_frags(sAhi, sAlo, mt, lane, ahi, alo);
      c[mt] = wmma3(ahi, alo, bhi, blo, c[mt]);
    }
    __syncthreads();
  }

  const int nc = n0 + wave * 16 + (lane & 15);
  const int lh = lane >> 4;
  const float bias = b1[e * HIDX + nc];
#pragma unroll
  for (int mt = 0; mt < 8; ++mt) {
#pragma unroll
    for (int v = 0; v < 8; ++v) {
      const int mloc = mt * 16 + v + 8 * lh;
      if (m0 + mloc < ne) {
        const int pid = srow[mloc];
        float val = c[mt][v] + bias;
        val = (val > 0.f ? val : 0.f) * topk_p[pid];
        const size_t hix = (size_t)pid * HIDX + nc;
        const __bf16 hb = (__bf16)val;
        hh[hix] = hb;
        hl[hix] = (__bf16)(val - (float)hb);
      }
    }
  }
}

// ---------------------------------------------------------------------------
// FAST FFN2: out[t, n] += h[pid] @ w2[e][:, n]
// ---------------------------------------------------------------------------
__global__ void __launch_bounds__(256)
ffn2_fast_kernel(const __bf16* __restrict__ hh, const __bf16* __restrict__ hl,
                 const __bf16* __restrict__ w2fh, const __bf16* __restrict__ w2fl,
                 const int* __restrict__ cnt, const int* __restrict__ pairlist,
                 float* __restrict__ out) {
  const int e  = blockIdx.z;
  const int ne = cnt[e];
  const int m0 = blockIdx.x * 128;
  if (m0 >= ne) return;
  const int n0 = blockIdx.y * 128;

  __shared__ __attribute__((aligned(32))) __bf16 sAhi[128 * 32];
  __shared__ __attribute__((aligned(32))) __bf16 sAlo[128 * 32];
  __shared__ int srow[128];

  const int tid = threadIdx.x, lane = tid & 31, wave = tid >> 5;
  for (int i = tid; i < 128; i += 256)
    srow[i] = pairlist[e * T_TOK + ((m0 + i) < ne ? (m0 + i) : 0)];
  __syncthreads();

  const v8f zero = {0.f, 0.f, 0.f, 0.f, 0.f, 0.f, 0.f, 0.f};
  v8f c[8];
#pragma unroll
  for (int i = 0; i < 8; ++i) c[i] = zero;

  const int NT = DIMX / 16, KC = HIDX / 32;
  const size_t bb = ((((size_t)(e * NT + (n0 >> 4) + wave)) * KC) * 32 + lane) * 16;

  for (int kc = 0; kc < KC; ++kc) {
    const int k0 = kc * 32;
    for (int i = tid; i < 512; i += 256) {
      const int m = i >> 2, k8 = i & 3;
      const size_t rb = (size_t)srow[m] * HIDX + k0 + k8 * 8;
      const int dst = A_DST(m, k8);
      *(v8bf*)(sAhi + dst) = *(const v8bf*)(hh + rb);
      *(v8bf*)(sAlo + dst) = *(const v8bf*)(hl + rb);
    }
    __syncthreads();
    const v16bf bhi = *(const v16bf*)(w2fh + bb + (size_t)kc * 512);
    const v16bf blo = *(const v16bf*)(w2fl + bb + (size_t)kc * 512);
#pragma unroll
    for (int mt = 0; mt < 8; ++mt) {
      v16bf ahi, alo;
      load_a_frags(sAhi, sAlo, mt, lane, ahi, alo);
      c[mt] = wmma3(ahi, alo, bhi, blo, c[mt]);
    }
    __syncthreads();
  }

  const int nc = n0 + wave * 16 + (lane & 15);
  const int lh = lane >> 4;
#pragma unroll
  for (int mt = 0; mt < 8; ++mt) {
#pragma unroll
    for (int v = 0; v < 8; ++v) {
      const int mloc = mt * 16 + v + 8 * lh;
      if (m0 + mloc < ne) {
        const int t = srow[mloc] >> 1;
        atomicAdd(&out[(size_t)t * DIMX + nc], c[mt][v]);
      }
    }
  }
}

// ===========================================================================
// FALLBACK PATH (smaller ws): conversion inside GEMM staging
// ===========================================================================
__global__ void __launch_bounds__(256)
ffn1_kernel(const float* __restrict__ x, const float* __restrict__ w1,
            const float* __restrict__ b1, const int* __restrict__ cnt,
            const int* __restrict__ pairlist, const float* __restrict__ topk_p,
            float* __restrict__ h) {
  const int e  = blockIdx.z;
  const int ne = cnt[e];
  const int m0 = blockIdx.x * 128;
  if (m0 >= ne) return;
  const int n0 = blockIdx.y * 128;

  __shared__ __attribute__((aligned(32))) __bf16 sAhi[128 * 32];
  __shared__ __attribute__((aligned(32))) __bf16 sAlo[128 * 32];
  __shared__ __attribute__((aligned(32))) __bf16 sBhi[32 * BLDS];
  __shared__ __attribute__((aligned(32))) __bf16 sBlo[32 * BLDS];
  __shared__ int srow[128];

  const int tid = threadIdx.x, lane = tid & 31, wave = tid >> 5;
  for (int i = tid; i < 128; i += 256)
    srow[i] = pairlist[e * T_TOK + ((m0 + i) < ne ? (m0 + i) : 0)];
  __syncthreads();

  const v8f zero = {0.f, 0.f, 0.f, 0.f, 0.f, 0.f, 0.f, 0.f};
  v8f c[8];
#pragma unroll
  for (int i = 0; i < 8; ++i) c[i] = zero;

  const float* wbase = w1 + (size_t)e * DIMX * HIDX;

  for (int k0 = 0; k0 < DIMX; k0 += 32) {
    for (int i = tid; i < 512; i += 256) {
      const int m = i >> 2, k8 = i & 3;
      const int tok = srow[m] >> 1;
      v8bf hv, lv;
      cvt8(x + (size_t)tok * DIMX + k0 + k8 * 8, hv, lv);
      const int dst = A_DST(m, k8);
      *(v8bf*)(sAhi + dst) = hv;
      *(v8bf*)(sAlo + dst) = lv;
    }
    for (int i = tid; i < 512; i += 256) {
      const int k = i >> 4, ng = i & 15;
      v8bf hv, lv;
      cvt8(wbase + (size_t)(k0 + k) * HIDX + n0 + ng * 8, hv, lv);
      *(v8bf*)(sBhi + k * BLDS + ng * 8) = hv;
      *(v8bf*)(sBlo + k * BLDS + ng * 8) = lv;
    }
    if (k0 + 32 < DIMX)
      __builtin_prefetch(wbase + (size_t)(k0 + 32 + (tid >> 3)) * HIDX + n0 + (tid & 7) * 16, 0, 0);
    __syncthreads();

    v16bf bhi, blo;
    load_b_frags_lds(sBhi, sBlo, wave * 16, lane, bhi, blo);
#pragma unroll
    for (int mt = 0; mt < 8; ++mt) {
      v16bf ahi, alo;
      load_a_frags(sAhi, sAlo, mt, lane, ahi, alo);
      c[mt] = wmma3(ahi, alo, bhi, blo, c[mt]);
    }
    __syncthreads();
  }

  const int nc = n0 + wave * 16 + (lane & 15);
  const int lh = lane >> 4;
  const float bias = b1[e * HIDX + nc];
#pragma unroll
  for (int mt = 0; mt < 8; ++mt) {
#pragma unroll
    for (int v = 0; v < 8; ++v) {
      const int mloc = mt * 16 + v + 8 * lh;
      if (m0 + mloc < ne) {
        const int pid = srow[mloc];
        float val = c[mt][v] + bias;
        val = val > 0.f ? val : 0.f;
        h[(size_t)pid * HIDX + nc] = val * topk_p[pid];
      }
    }
  }
}

__global__ void __launch_bounds__(256)
ffn2_kernel(const float* __restrict__ h, const float* __restrict__ w2,
            const int* __restrict__ cnt, const int* __restrict__ pairlist,
            float* __restrict__ out) {
  const int e  = blockIdx.z;
  const int ne = cnt[e];
  const int m0 = blockIdx.x * 128;
  if (m0 >= ne) return;
  const int n0 = blockIdx.y * 128;

  __shared__ __attribute__((aligned(32))) __bf16 sAhi[128 * 32];
  __shared__ __attribute__((aligned(32))) __bf16 sAlo[128 * 32];
  __shared__ __attribute__((aligned(32))) __bf16 sBhi[32 * BLDS];
  __shared__ __attribute__((aligned(32))) __bf16 sBlo[32 * BLDS];
  __shared__ int srow[128];

  const int tid = threadIdx.x, lane = tid & 31, wave = tid >> 5;
  for (int i = tid; i < 128; i += 256)
    srow[i] = pairlist[e * T_TOK + ((m0 + i) < ne ? (m0 + i) : 0)];
  __syncthreads();

  const v8f zero = {0.f, 0.f, 0.f, 0.f, 0.f, 0.f, 0.f, 0.f};
  v8f c[8];
#pragma unroll
  for (int i = 0; i < 8; ++i) c[i] = zero;

  const float* wbase = w2 + (size_t)e * HIDX * DIMX;

  for (int k0 = 0; k0 < HIDX; k0 += 32) {
    for (int i = tid; i < 512; i += 256) {
      const int m = i >> 2, k8 = i & 3;
      v8bf hv, lv;
      cvt8(h + (size_t)srow[m] * HIDX + k0 + k8 * 8, hv, lv);
      const int dst = A_DST(m, k8);
      *(v8bf*)(sAhi + dst) = hv;
      *(v8bf*)(sAlo + dst) = lv;
    }
    for (int i = tid; i < 512; i += 256) {
      const int k = i >> 4, ng = i & 15;
      v8bf hv, lv;
      cvt8(wbase + (size_t)(k0 + k) * DIMX + n0 + ng * 8, hv, lv);
      *(v8bf*)(sBhi + k * BLDS + ng * 8) = hv;
      *(v8bf*)(sBlo + k * BLDS + ng * 8) = lv;
    }
    if (k0 + 32 < HIDX)
      __builtin_prefetch(wbase + (size_t)(k0 + 32 + (tid >> 3)) * DIMX + n0 + (tid & 7) * 16, 0, 0);
    __syncthreads();

    v16bf bhi, blo;
    load_b_frags_lds(sBhi, sBlo, wave * 16, lane, bhi, blo);
#pragma unroll
    for (int mt = 0; mt < 8; ++mt) {
      v16bf ahi, alo;
      load_a_frags(sAhi, sAlo, mt, lane, ahi, alo);
      c[mt] = wmma3(ahi, alo, bhi, blo, c[mt]);
    }
    __syncthreads();
  }

  const int nc = n0 + wave * 16 + (lane & 15);
  const int lh = lane >> 4;
#pragma unroll
  for (int mt = 0; mt < 8; ++mt) {
#pragma unroll
    for (int v = 0; v < 8; ++v) {
      const int mloc = mt * 16 + v + 8 * lh;
      if (m0 + mloc < ne) {
        const int t = srow[mloc] >> 1;
        atomicAdd(&out[(size_t)t * DIMX + nc], c[mt][v]);
      }
    }
  }
}

// ---------------------------------------------------------------------------
// launch
// ---------------------------------------------------------------------------
extern "C" void kernel_launch(void* const* d_in, const int* in_sizes, int n_in,
                              void* d_out, int out_size, void* d_ws, size_t ws_size,
                              hipStream_t stream) {
  (void)in_sizes; (void)n_in; (void)out_size;
  const float* x  = (const float*)d_in[0];
  const float* gw = (const float*)d_in[1];
  const float* gb = (const float*)d_in[2];
  const float* w1 = (const float*)d_in[3];
  const float* b1 = (const float*)d_in[4];
  const float* w2 = (const float*)d_in[5];
  const float* b2 = (const float*)d_in[6];
  float* out = (float*)d_out;

  char* ws = (char*)d_ws;
  const size_t BASE = 196864;                // cnt/topk/pairlist block
  int*   cnt      = (int*)(ws);
  int*   topk_e   = (int*)(ws + 256);
  float* topk_p   = (float*)(ws + 256 + 32768);
  int*   pairlist = (int*)(ws + 256 + 65536);

  const size_t SZ_HPLANE = (size_t)2 * T_TOK * HIDX * 2;      // 67,108,864 B
  const size_t SZ_XPLANE = (size_t)T_TOK * DIMX * 2;          //  8,388,608 B
  const size_t SZ_WPLANE = (size_t)NEXP * DIMX * HIDX * 2;    // 67,108,864 B
  const size_t FAST_NEED = BASE + 2 * SZ_HPLANE + 2 * SZ_XPLANE + 4 * SZ_WPLANE;

  init_cnt_kernel<<<1, 32, 0, stream>>>(cnt);
  gate_kernel<<<T_TOK / 256, 256, 0, stream>>>(x, gw, gb, cnt, pairlist, topk_e, topk_p);
  out_init_kernel<<<T_TOK, 256, 0, stream>>>(b2, topk_e, topk_p, out);

  if (ws_size >= FAST_NEED) {
    size_t off = BASE;
    __bf16* hh   = (__bf16*)(ws + off); off += SZ_HPLANE;
    __bf16* hl   = (__bf16*)(ws + off); off += SZ_HPLANE;
    __bf16* xh   = (__bf16*)(ws + off); off += SZ_XPLANE;
    __bf16* xl   = (__bf16*)(ws + off); off += SZ_XPLANE;
    __bf16* w1fh = (__bf16*)(ws + off); off += SZ_WPLANE;
    __bf16* w1fl = (__bf16*)(ws + off); off += SZ_WPLANE;
    __bf16* w2fh = (__bf16*)(ws + off); off += SZ_WPLANE;
    __bf16* w2fl = (__bf16*)(ws + off);

    cvt_x_kernel<<<(T_TOK * DIMX / 8) / 256, 256, 0, stream>>>(x, xh, xl);
    cvt_w_kernel<<<(NEXP * (DIMX / 16) * HIDX) / 256, 256, 0, stream>>>(w1, w1fh, w1fl, DIMX, HIDX);
    cvt_w_kernel<<<(NEXP * (HIDX / 16) * DIMX) / 256, 256, 0, stream>>>(w2, w2fh, w2fl, HIDX, DIMX);

    ffn1_fast_kernel<<<dim3(T_TOK / 128, HIDX / 128, NEXP), 256, 0, stream>>>(
        xh, xl, w1fh, w1fl, b1, cnt, pairlist, topk_p, hh, hl);
    ffn2_fast_kernel<<<dim3(T_TOK / 128, DIMX / 128, NEXP), 256, 0, stream>>>(
        hh, hl, w2fh, w2fl, cnt, pairlist, out);
  } else {
    float* h = (float*)(ws + BASE);          // 134 MB fp32 h
    ffn1_kernel<<<dim3(T_TOK / 128, HIDX / 128, NEXP), 256, 0, stream>>>(
        x, w1, b1, cnt, pairlist, topk_p, h);
    ffn2_kernel<<<dim3(T_TOK / 128, DIMX / 128, NEXP), 256, 0, stream>>>(
        h, w2, cnt, pairlist, out);
  }
}
